// Wav2Vec2GumbelVectorQuantizer_55336358642981
// MI455X (gfx1250) — matrix operand order repaired and checked by reading, not compiled, not run on hardware
//
#include <hip/hip_runtime.h>
#include <hip/hip_bf16.h>

// Shapes (fixed by setup_inputs): B=8,S=4096,H=512, G=2,V=320, Dg=128
// M = B*S = 32768 rows, N = G*V = 640 cols, K = H = 512.
#define MROWS   32768
#define KDIM    512
#define NCOLS   640
#define GRP     2
#define VSZ     320
#define DG      128
#define NTILES  40          // 640/16
#define GTILES  20          // 320/16
#define KSTEPS  16          // 512/32
#define BPAD    520         // 512 + 8 halves padding (bank-conflict dodge)
#define ROWS_PER_BLOCK 128  // 8 waves * 16 rows

typedef __attribute__((ext_vector_type(16))) __bf16 v16bf;
typedef __attribute__((ext_vector_type(2)))  __bf16 bf16x2;
typedef __attribute__((ext_vector_type(2)))  float  f32x2;
typedef __attribute__((ext_vector_type(8)))  float  v8f;

struct BF16x16 { uint4 lo; uint4 hi; };   // 32 bytes == v16bf

// Packed f32->bf16 (RNE) pair: backend emits v_cvt_pk_bf16_f32.
__device__ __forceinline__ unsigned cvt2(float a, float b) {
    f32x2 f = {a, b};
    bf16x2 h = __builtin_convertvector(f, bf16x2);
    return __builtin_bit_cast(unsigned, h);
}

// B fragment from LDS: two adjacent 16B chunks (16 consecutive K halves).
__device__ __forceinline__ v16bf ldsB_frag(const unsigned short* p) {
    BF16x16 t;
    t.lo = *(const uint4*)p;              // ds_load_b128
    t.hi = *(const uint4*)(p + 8);        // ds_load_b128
    return __builtin_bit_cast(v16bf, t);
}

// Fused: logits GEMM (bf16 WMMA, A resident in 128 VGPRs for full K) +
// per-group argmax + codevector gather + histogram.
__global__ __launch_bounds__(256)
void vq_fused_kernel(const float* __restrict__ hs,
                     const float* __restrict__ Wp,
                     const float* __restrict__ bp,
                     const float* __restrict__ cb,
                     float* __restrict__ outp,
                     int* __restrict__ cnt) {
    // Double-buffered B tile only: 2 * 16 cols * 520 halves * 2B = 33,280 B
    __shared__ __align__(16) unsigned short lds_b[2 * 16 * BPAD];

    const int tid  = threadIdx.x;
    const int w    = tid >> 5;
    const int lane = tid & 31;
    const int m0   = blockIdx.x * ROWS_PER_BLOCK;

    // ---- Load A fragments for all 16 k-steps straight from global (once).
    // A 16x32 layout (matches B): lane<16 -> row=lane, K {0..15};
    // lane>=16 -> row=lane-16, K {16..31}. Lane r + lane r+16 together
    // cover one full 128B line of hs per k-step.
    v16bf afrag[KSTEPS];
    {
        const float* hsrow = hs + (size_t)(m0 + w * 16 + (lane & 15)) * KDIM
                                + ((lane < 16) ? 0 : 16);
        #pragma unroll
        for (int ks = 0; ks < KSTEPS; ++ks) {
            float4 x0 = *(const float4*)(hsrow + ks * 32 + 0);
            float4 x1 = *(const float4*)(hsrow + ks * 32 + 4);
            float4 x2 = *(const float4*)(hsrow + ks * 32 + 8);
            float4 x3 = *(const float4*)(hsrow + ks * 32 + 12);
            BF16x16 t;
            t.lo.x = cvt2(x0.x, x0.y); t.lo.y = cvt2(x0.z, x0.w);
            t.lo.z = cvt2(x1.x, x1.y); t.lo.w = cvt2(x1.z, x1.w);
            t.hi.x = cvt2(x2.x, x2.y); t.hi.y = cvt2(x2.z, x2.w);
            t.hi.z = cvt2(x3.x, x3.y); t.hi.w = cvt2(x3.z, x3.w);
            afrag[ks] = __builtin_bit_cast(v16bf, t);
        }
    }

    // ---- B stager: coalesced float4 row loads + 4x4 register transpose ->
    // packed bf16 column stores (column-major, padded).
    const float4* W4 = (const float4*)Wp;
    auto stageB = [&](int nt, int buf) {
        unsigned short* dst = &lds_b[buf * 16 * BPAD];
        #pragma unroll
        for (int it = 0; it < 2; ++it) {
            int mb = tid + it * 256;          // 512 micro-blocks of 4k x 4n
            int kb = mb >> 2, nb = mb & 3;
            int k0 = kb * 4, c0 = nb * 4;
            float4 r0 = W4[(size_t)(k0 + 0) * (NCOLS / 4) + nt * 4 + nb];
            float4 r1 = W4[(size_t)(k0 + 1) * (NCOLS / 4) + nt * 4 + nb];
            float4 r2 = W4[(size_t)(k0 + 2) * (NCOLS / 4) + nt * 4 + nb];
            float4 r3 = W4[(size_t)(k0 + 3) * (NCOLS / 4) + nt * 4 + nb];
            uint2 z;
            z.x = cvt2(r0.x, r1.x); z.y = cvt2(r2.x, r3.x);
            *(uint2*)&dst[(c0 + 0) * BPAD + k0] = z;
            z.x = cvt2(r0.y, r1.y); z.y = cvt2(r2.y, r3.y);
            *(uint2*)&dst[(c0 + 1) * BPAD + k0] = z;
            z.x = cvt2(r0.z, r1.z); z.y = cvt2(r2.z, r3.z);
            *(uint2*)&dst[(c0 + 2) * BPAD + k0] = z;
            z.x = cvt2(r0.w, r1.w); z.y = cvt2(r2.w, r3.w);
            *(uint2*)&dst[(c0 + 3) * BPAD + k0] = z;
        }
    };

    stageB(0, 0);
    __syncthreads();

    const int bhalf = (lane < 16) ? 0 : 16;   // B 32x16: lane<16 K0..15, else K16..31

    float rbest[8];
    int   rbidx[8];
    #pragma unroll
    for (int v = 0; v < 8; ++v) { rbest[v] = -3.402823466e+38f; rbidx[v] = 0; }

    #pragma unroll 1
    for (int nt = 0; nt < NTILES; ++nt) {
        const int p = nt & 1;
        if (nt + 1 < NTILES) stageB(nt + 1, p ^ 1);   // overlap with WMMAs

        // ---- 16x16 logit tile: 16 chained bf16 WMMAs. B fragments flow
        // through a 4-deep rotating register ring loaded 4 steps ahead.
        const unsigned short* bcol =
            &lds_b[p * 16 * BPAD + (lane & 15) * BPAD + bhalf];
        v16bf bb[4];
        #pragma unroll
        for (int i = 0; i < 4; ++i) bb[i] = ldsB_frag(bcol + i * 32);
        v8f c = {};
        #pragma unroll
        for (int ks = 0; ks < KSTEPS; ++ks) {
            c = __builtin_amdgcn_wmma_f32_16x16x32_bf16(
                    false, afrag[ks], false, bb[ks & 3], (short)0, c,
                    false, false);
            if (ks + 4 < KSTEPS) bb[ks & 3] = ldsB_frag(bcol + (ks + 4) * 32);
        }
        // Pin the schedule: 8 DS reads (ring fill), then 12x(WMMA, 2 DS
        // reads), then 4 tail WMMAs. Forces each WMMA to wait with ~6
        // ds_loads outstanding (s_wait_dscnt 6) instead of 0, and keeps the
        // 4 ring slots in distinct registers. 0x8 = MFMA/WMMA, 0x100 = DS rd.
        __builtin_amdgcn_sched_group_barrier(0x100, 8, 0);
        #pragma unroll
        for (int ks = 0; ks < 12; ++ks) {
            __builtin_amdgcn_sched_group_barrier(0x008, 1, 0);
            __builtin_amdgcn_sched_group_barrier(0x100, 2, 0);
        }
        __builtin_amdgcn_sched_group_barrier(0x008, 4, 0);

        // ---- per-row argmax update (first-index tie-break, like jnp.argmax)
        const float bias = bp[nt * 16 + (lane & 15)];
        const int   colg = ((nt % GTILES) << 4) | (lane & 15);
        #pragma unroll
        for (int v = 0; v < 8; ++v) {
            float val = c[v] + bias;
            int   idx = colg;
            #pragma unroll
            for (int m = 1; m <= 8; m <<= 1) {   // stays within 16-lane halves
                float ov = __shfl_xor(val, m, 32);
                int   oi = __shfl_xor(idx, m, 32);
                if (ov > val || (ov == val && oi < idx)) { val = ov; idx = oi; }
            }
            if (val > rbest[v]) { rbest[v] = val; rbidx[v] = idx; }
        }

        // ---- group boundary: gather codevectors, histogram, reset ----
        if ((nt % GTILES) == (GTILES - 1)) {
            const int g = nt / GTILES;
            #pragma unroll
            for (int r = 0; r < 16; ++r) {
                int id  = __shfl(rbidx[r & 7], (r & 8) ? 16 : 0, 32);
                size_t row = (size_t)(m0 + w * 16 + r);
                // 128 floats: each lane moves one float4
                float4 cv = *(const float4*)(cb + ((size_t)(g * VSZ + id) * DG)
                                                + lane * 4);
                *(float4*)(outp + row * (GRP * DG) + g * DG + lane * 4) = cv;
                if (lane == 0) atomicAdd(&cnt[g * VSZ + id], 1);
            }
            #pragma unroll
            for (int v = 0; v < 8; ++v) {
                rbest[v] = -3.402823466e+38f; rbidx[v] = 0;
            }
        }
        __syncthreads();   // buf[p] free for restage; buf[p^1] fully staged
    }
}

__global__ void zero_counts_kernel(int* __restrict__ cnt) {
    int i = threadIdx.x;
    if (i < NCOLS) cnt[i] = 0;
}

__global__ void perplexity_kernel(const int* __restrict__ cnt,
                                  float* __restrict__ out_ppl) {
    __shared__ float terms[NCOLS];
    int i = threadIdx.x;
    if (i < NCOLS) {
        float m = (float)cnt[i] * (1.0f / (float)MROWS);
        terms[i] = m * logf(m + 1e-7f);
    }
    __syncthreads();
    if (i == 0) {
        float p = 0.f;
        for (int g = 0; g < GRP; ++g) {
            float s = 0.f;
            for (int v = 0; v < VSZ; ++v) s += terms[g * VSZ + v];
            p += expf(-s);
        }
        out_ppl[0] = p;
    }
}

extern "C" void kernel_launch(void* const* d_in, const int* in_sizes, int n_in,
                              void* d_out, int out_size, void* d_ws, size_t ws_size,
                              hipStream_t stream) {
    const float* hs = (const float*)d_in[0];   // (8,4096,512)  f32
    const float* Wp = (const float*)d_in[1];   // (512,640)     f32
    const float* bp = (const float*)d_in[2];   // (640,)        f32
    const float* cb = (const float*)d_in[3];   // (1,640,128)   f32
    // d_in[4] = num_groups (==2, hardcoded in tiling)

    float* outp    = (float*)d_out;                 // (8,4096,256) flat
    float* out_ppl = (float*)d_out + (size_t)MROWS * GRP * DG; // scalar
    int*   cnt     = (int*)d_ws;                    // 640 ints

    zero_counts_kernel<<<1, NCOLS, 0, stream>>>(cnt);
    vq_fused_kernel<<<MROWS / ROWS_PER_BLOCK, 256, 0, stream>>>(
        hs, Wp, bp, cb, outp, cnt);
    perplexity_kernel<<<1, NCOLS, 0, stream>>>(cnt, out_ppl);
}
